// MLA_26353919328640
// MI455X (gfx1250) — matrix operand order repaired and checked by reading, not compiled, not run on hardware
//
#include <hip/hip_runtime.h>
#include <math.h>
#include <stddef.h>
#include <stdint.h>

// ---------------- constants ----------------
#define B_ 2
#define S_ 2048
#define D_ 2048
#define H_ 16
#define KVH_ 8
#define HD_ 128
#define KVR_ 512
#define BS_ (B_ * S_)
#define NEGBIG (-1.0e9f)
#define LN10000 9.210340371976184f

typedef __bf16 bf16_t;
typedef bf16_t bf16x8 __attribute__((ext_vector_type(8)));
typedef bf16_t bf16x16 __attribute__((ext_vector_type(16)));
typedef float f32x8 __attribute__((ext_vector_type(8)));

// ---------------- gfx1250 async memory->LDS path ----------------
#if __has_builtin(__builtin_amdgcn_global_load_async_to_lds_b128) && \
    __has_builtin(__builtin_amdgcn_s_wait_asynccnt)
#define HAS_ASYNC_LDS 1
#else
#define HAS_ASYNC_LDS 0
#endif

// 16-byte memory -> LDS copy. Async (ASYNCcnt) on gfx1250, sync fallback else.
// Builtin signature (from hipcc diagnostic): first param is
// 'int __attribute__((vector_size(16))) __device__ *' (global side), then LDS
// pointer, then imm offset and imm cpol.
__device__ inline void async_cp16(const void* g, void* l) {
#if HAS_ASYNC_LDS
  typedef int v4i_ __attribute__((vector_size(16)));
  __builtin_amdgcn_global_load_async_to_lds_b128(
      (__attribute__((address_space(1))) v4i_*)(uintptr_t)g,
      (__attribute__((address_space(3))) v4i_*)(uint32_t)(uintptr_t)l, 0, 0);
#else
  *(bf16x8*)l = *(const bf16x8*)g;
#endif
}

template <unsigned short N>
__device__ inline void wait_async() {
#if HAS_ASYNC_LDS
  __builtin_amdgcn_s_wait_asynccnt(N);
#endif
}

// ---------------- WMMA helpers ----------------
// A-fragment (16x32 bf16, row-major source [M][K]):
// lanes 0-15: M=lane, K {base..base+7, base+16..base+23}
// lanes 16-31: M=lane-16, K {base+8..base+15, base+24..base+31}
__device__ inline bf16x16 load_fragA(const bf16_t* base, size_t stride) {
  const int lane = threadIdx.x & 31;
  const int lr = lane & 15;
  const int hf = lane >> 4;
  const bf16_t* p = base + (size_t)lr * stride + hf * 8;
  bf16x8 lo = *(const bf16x8*)p;
  bf16x8 hi = *(const bf16x8*)(p + 16);
  bf16x16 r;
#pragma unroll
  for (int i = 0; i < 8; ++i) { r[i] = lo[i]; r[i + 8] = hi[i]; }
  return r;
}

// B-fragment (32x16 bf16) from N-major storage [N][K]:
// lanes 0-15: N=lane, K {base..base+7, base+16..base+23} ; lanes 16-31: N=lane-16, K {base+8..base+15, base+24..base+31}
__device__ inline bf16x16 load_fragB(const bf16_t* base, size_t stride) {
  const int lane = threadIdx.x & 31;
  const int lr = lane & 15;
  const int hf = lane >> 4;
  const bf16_t* p = base + (size_t)lr * stride + hf * 8;
  bf16x8 lo = *(const bf16x8*)p;
  bf16x8 hi = *(const bf16x8*)(p + 16);
  bf16x16 r;
#pragma unroll
  for (int i = 0; i < 8; ++i) { r[i] = lo[i]; r[i + 8] = hi[i]; }
  return r;
}

__device__ inline f32x8 wmma_bf16(bf16x16 a, bf16x16 b, f32x8 c) {
  return __builtin_amdgcn_wmma_f32_16x16x32_bf16(false, a, false, b, (short)0, c,
                                                 false, false);
}

__device__ inline f32x8 f32x8_zero() {
  f32x8 z = {0.f, 0.f, 0.f, 0.f, 0.f, 0.f, 0.f, 0.f};
  return z;
}

// ---------------- elementwise converters ----------------
__global__ __launch_bounds__(256) void conv_bf16_kernel(const float* __restrict__ in,
                                                        bf16_t* __restrict__ out,
                                                        long long n) {
  long long i = (long long)blockIdx.x * 256 + threadIdx.x;
  if (i < n) out[i] = (bf16_t)in[i];
}

// in: [K][N] f32 row-major -> out: [N][K] bf16 (B^T for GEMM)
__global__ __launch_bounds__(256) void transconv_kernel(const float* __restrict__ in,
                                                        bf16_t* __restrict__ out,
                                                        int Kd, int Nd) {
  long long i = (long long)blockIdx.x * 256 + threadIdx.x;
  if (i < (long long)Kd * Nd) {
    int n = (int)(i / Kd);
    int k = (int)(i % Kd);
    out[i] = (bf16_t)in[(size_t)k * Nd + n];
  }
}

// ---------------- generic WMMA GEMM ----------------
// C[MxN] (OutT) = A[MxK] (bf16 row-major) * B (given as BT[NxK] bf16)
// block tile 128x128; 8 waves: 4(M) x 2(N), wave tile 32x64.
// Double-buffered K-stage of 32 with async memory->LDS copies.
template <typename OutT>
__global__ __launch_bounds__(256) void gemm_kernel(const bf16_t* __restrict__ A,
                                                   const bf16_t* __restrict__ BT,
                                                   OutT* __restrict__ C,
                                                   int M, int N, int K) {
  __shared__ bf16_t sA[2][128][40];  // 80B rows: 16B aligned, conflict-free
  __shared__ bf16_t sB[2][128][40];
  const int tid = threadIdx.x;
  const int wid = tid >> 5;
  const int lane = tid & 31;
  const int lcol = lane & 15;
  const int half = lane >> 4;
  const int mBase = blockIdx.x * 128;
  const int nBase = blockIdx.y * 128;
  const int mw = (wid & 3) * 32;
  const int nw = (wid >> 2) * 64;

  f32x8 acc[2][4];
#pragma unroll
  for (int mt = 0; mt < 2; ++mt)
#pragma unroll
    for (int nt = 0; nt < 4; ++nt) acc[mt][nt] = f32x8_zero();

  // 4 async b128 issues per thread per stage (512 chunks per matrix / 256 thr)
  auto issue_stage = [&](int buf, int k0) {
#pragma unroll
    for (int i = 0; i < 2; ++i) {
      int cc = tid + i * 256;
      int r = cc >> 2;
      int c8 = (cc & 3) * 8;
      async_cp16(&A[(size_t)(mBase + r) * K + k0 + c8], &sA[buf][r][c8]);
      async_cp16(&BT[(size_t)(nBase + r) * K + k0 + c8], &sB[buf][r][c8]);
    }
  };
  auto compute_stage = [&](int buf) {
    bf16x16 a0 = load_fragA(&sA[buf][mw][0], 40);
    bf16x16 a1 = load_fragA(&sA[buf][mw + 16][0], 40);
#pragma unroll
    for (int nt = 0; nt < 4; ++nt) {
      bf16x16 bf = load_fragB(&sB[buf][nw + nt * 16][0], 40);
      acc[0][nt] = wmma_bf16(a0, bf, acc[0][nt]);
      acc[1][nt] = wmma_bf16(a1, bf, acc[1][nt]);
    }
  };

  const int nk = K >> 5;
  issue_stage(0, 0);
  for (int k = 0; k < nk; ++k) {
    if (k + 1 < nk) {
      issue_stage((k + 1) & 1, (k + 1) * 32);  // prefetch next tile (other buffer)
      wait_async<4>();                         // tile k landed (next 4 in flight)
    } else {
      wait_async<0>();
    }
    __syncthreads();        // all waves' tile-k data visible
    compute_stage(k & 1);
    __syncthreads();        // done reading before iter k+2 overwrites this buffer
  }

#pragma unroll
  for (int mt = 0; mt < 2; ++mt)
#pragma unroll
    for (int nt = 0; nt < 4; ++nt) {
      int col = nBase + nw + nt * 16 + lcol;
      int row0 = mBase + mw + mt * 16 + half * 8;
#pragma unroll
      for (int i = 0; i < 8; ++i)
        C[(size_t)(row0 + i) * N + col] = (OutT)acc[mt][nt][i];
    }
}

// ---------------- RoPE on q + relayout [BS][H*256] -> [B][H][S][256] ----------------
__global__ __launch_bounds__(256) void ropeq_kernel(const bf16_t* __restrict__ qraw,
                                                    const int* __restrict__ pos,
                                                    bf16_t* __restrict__ qb) {
  int idx = blockIdx.x * 256 + threadIdx.x;  // BS*H*128 pairs, exact
  int pr = idx & 127;
  int h = (idx >> 7) & (H_ - 1);
  int rf = idx >> 11;          // 0..BS-1
  int s = rf & (S_ - 1);
  int b = rf >> 11;
  const bf16_t* src = qraw + (size_t)rf * (H_ * 2 * HD_) + h * 256;
  bf16_t* dst = qb + ((size_t)(b * H_ + h) * S_ + s) * 256;
  if (pr < 64) {  // nope half: copy
    dst[2 * pr] = src[2 * pr];
    dst[2 * pr + 1] = src[2 * pr + 1];
  } else {        // pe half: rotate
    int p = pr - 64;
    float xr = (float)src[128 + 2 * p];
    float xi = (float)src[128 + 2 * p + 1];
    float freq = 40.0f * __expf(-((float)(2 * p) / 128.0f) * LN10000);
    float ang = (float)pos[s] * freq;
    float c = __cosf(ang), sn = __sinf(ang);
    dst[128 + 2 * p] = (bf16_t)(xr * c - xi * sn);
    dst[128 + 2 * p + 1] = (bf16_t)(xr * sn + xi * c);
  }
}

// ---------------- layernorm(kv_c) + rope(k_pe); one block per token ----------------
__global__ __launch_bounds__(256) void kvproc_kernel(const float* __restrict__ kv,
                                                     const float* __restrict__ gamma,
                                                     const int* __restrict__ pos,
                                                     bf16_t* __restrict__ kvc,
                                                     bf16_t* __restrict__ kpe) {
  __shared__ float red[256];
  const int row = blockIdx.x;  // 0..BS-1
  const int tid = threadIdx.x;
  const float* x = kv + (size_t)row * (KVR_ + HD_);
  float x0 = x[tid], x1 = x[tid + 256];
  red[tid] = x0 + x1;
  __syncthreads();
  for (int off = 128; off > 0; off >>= 1) {
    if (tid < off) red[tid] += red[tid + off];
    __syncthreads();
  }
  float mean = red[0] / 512.0f;
  __syncthreads();
  float d0 = x0 - mean, d1 = x1 - mean;
  red[tid] = d0 * d0 + d1 * d1;
  __syncthreads();
  for (int off = 128; off > 0; off >>= 1) {
    if (tid < off) red[tid] += red[tid + off];
    __syncthreads();
  }
  float rstd = rsqrtf(red[0] / 512.0f + 1e-5f);
  kvc[(size_t)row * KVR_ + tid] = (bf16_t)(d0 * rstd * gamma[tid]);
  kvc[(size_t)row * KVR_ + tid + 256] = (bf16_t)(d1 * rstd * gamma[tid + 256]);
  if (tid < 64) {
    int s = row & (S_ - 1);
    int p = tid;
    float xr = x[512 + 2 * p], xi = x[512 + 2 * p + 1];
    float freq = 40.0f * __expf(-((float)(2 * p) / 128.0f) * LN10000);
    float ang = (float)pos[s] * freq;
    float c = __cosf(ang), sn = __sinf(ang);
    kpe[(size_t)row * 128 + 2 * p] = (bf16_t)(xr * c - xi * sn);
    kpe[(size_t)row * 128 + 2 * p + 1] = (bf16_t)(xr * sn + xi * c);
  }
}

// ---------------- assemble K [B][KVH][S][256] and V^T [B][KVH][128][S] ----------------
__global__ __launch_bounds__(256) void buildkv_kernel(const bf16_t* __restrict__ kvb,
                                                      const bf16_t* __restrict__ kpe,
                                                      bf16_t* __restrict__ Kb,
                                                      bf16_t* __restrict__ VT) {
  int idx = blockIdx.x * 256 + threadIdx.x;  // B*S*KVH*128, exact
  int d = idx & 127;
  int kvh = (idx >> 7) & (KVH_ - 1);
  int rf = idx >> 10;  // 0..BS-1
  int s = rf & (S_ - 1);
  int b = rf >> 11;
  const bf16_t* src = kvb + (size_t)rf * (KVH_ * 2 * HD_) + kvh * 256;
  size_t ko = ((size_t)(b * KVH_ + kvh) * S_ + s) * 256;
  Kb[ko + d] = src[d];                                  // k_nope
  Kb[ko + 128 + d] = kpe[(size_t)rf * 128 + d];         // k_pe broadcast
  VT[((size_t)(b * KVH_ + kvh) * 128 + d) * S_ + s] = src[128 + d];
}

// ---------------- flash attention ----------------
// grid (S/128, H, B); 8 waves, wave owns 16 q rows; kk tiles of 64.
__global__ __launch_bounds__(256) void attn_kernel(const bf16_t* __restrict__ Q,
                                                   const bf16_t* __restrict__ Kb,
                                                   const bf16_t* __restrict__ VT,
                                                   bf16_t* __restrict__ Ob,
                                                   float scale) {
  __shared__ bf16_t sK[64][264];   // 528B rows: 16B aligned, conflict-free
  __shared__ bf16_t sP[128][88];
  const int tid = threadIdx.x;
  const int wid = tid >> 5;
  const int lane = tid & 31;
  const int lcol = lane & 15;
  const int half = lane >> 4;
  const int qt = blockIdx.x, h = blockIdx.y, b = blockIdx.z;
  const int kvh = h >> 1;  // rep = H/KVH = 2
  const int qbase = qt * 128;
  const int wq0 = qbase + wid * 16;

  // resident Q fragments (16 rows x 256)
  const bf16_t* qptr = Q + ((size_t)(b * H_ + h) * S_ + wq0) * 256;
  bf16x16 qf[8];
#pragma unroll
  for (int ks = 0; ks < 8; ++ks) qf[ks] = load_fragA(qptr + ks * 32, 256);

  f32x8 O[8];
  float rmax[8], rsum[8];
#pragma unroll
  for (int i = 0; i < 8; ++i) { O[i] = f32x8_zero(); rmax[i] = -INFINITY; rsum[i] = 0.f; }

  const bf16_t* kp = Kb + (size_t)(b * KVH_ + kvh) * S_ * 256;
  const bf16_t* vp = VT + (size_t)(b * KVH_ + kvh) * 128 * S_;
  const int ktiles = (qbase + 128) >> 6;

  for (int kt = 0; kt < ktiles; ++kt) {
    const int kk0 = kt * 64;
    // stage K tile 64x256 via async memory->LDS
#pragma unroll
    for (int i = 0; i < 8; ++i) {
      int cc = tid + i * 256;
      int r = cc >> 5;
      int c8 = (cc & 31) * 8;
      async_cp16(&kp[(size_t)(kk0 + r) * 256 + c8], &sK[r][c8]);
    }
    // warm L2/WGP$ for next K and V tiles while this tile is in flight
    if (kt + 1 < ktiles) {
      const int nk0 = kk0 + 64;
      __builtin_prefetch(&kp[(size_t)(nk0 + (tid & 63)) * 256 + (tid >> 6) * 64], 0, 0);
      __builtin_prefetch(&vp[(size_t)(tid & 127) * S_ + nk0 + (tid >> 7) * 32], 0, 0);
    }
    wait_async<0>();
    __syncthreads();

    // scores: 16 x 64, K=256
    f32x8 st[4];
#pragma unroll
    for (int nt = 0; nt < 4; ++nt) st[nt] = f32x8_zero();
#pragma unroll
    for (int ks = 0; ks < 8; ++ks) {
#pragma unroll
      for (int nt = 0; nt < 4; ++nt) {
        bf16x16 bf = load_fragB(&sK[nt * 16][ks * 32], 264);
        st[nt] = wmma_bf16(qf[ks], bf, st[nt]);
      }
    }

    // scale + causal mask + row max
    float tmax[8];
#pragma unroll
    for (int i = 0; i < 8; ++i) tmax[i] = -INFINITY;
#pragma unroll
    for (int nt = 0; nt < 4; ++nt) {
      int col = kk0 + nt * 16 + lcol;
#pragma unroll
      for (int i = 0; i < 8; ++i) {
        int row = wq0 + half * 8 + i;
        float v = st[nt][i] * scale + (col > row ? NEGBIG : 0.0f);
        st[nt][i] = v;
        tmax[i] = fmaxf(tmax[i], v);
      }
    }
#pragma unroll
    for (int i = 0; i < 8; ++i) {
      tmax[i] = fmaxf(tmax[i], __shfl_xor(tmax[i], 1, 32));
      tmax[i] = fmaxf(tmax[i], __shfl_xor(tmax[i], 2, 32));
      tmax[i] = fmaxf(tmax[i], __shfl_xor(tmax[i], 4, 32));
      tmax[i] = fmaxf(tmax[i], __shfl_xor(tmax[i], 8, 32));
    }
    float alpha[8], tsum[8];
#pragma unroll
    for (int i = 0; i < 8; ++i) {
      float mn = fmaxf(rmax[i], tmax[i]);
      alpha[i] = __expf(rmax[i] - mn);
      rmax[i] = mn;
      tsum[i] = 0.f;
    }
#pragma unroll
    for (int nt = 0; nt < 4; ++nt)
#pragma unroll
      for (int i = 0; i < 8; ++i) {
        float p = __expf(st[nt][i] - rmax[i]);
        st[nt][i] = p;
        tsum[i] += p;
      }
#pragma unroll
    for (int i = 0; i < 8; ++i) {
      tsum[i] += __shfl_xor(tsum[i], 1, 32);
      tsum[i] += __shfl_xor(tsum[i], 2, 32);
      tsum[i] += __shfl_xor(tsum[i], 4, 32);
      tsum[i] += __shfl_xor(tsum[i], 8, 32);
      rsum[i] = rsum[i] * alpha[i] + tsum[i];
    }
#pragma unroll
    for (int nt = 0; nt < 8; ++nt)
#pragma unroll
      for (int i = 0; i < 8; ++i) O[nt][i] *= alpha[i];

    // P -> LDS (bf16) to re-fragment as A operand
#pragma unroll
    for (int nt = 0; nt < 4; ++nt)
#pragma unroll
      for (int i = 0; i < 8; ++i)
        sP[wid * 16 + half * 8 + i][nt * 16 + lcol] = (bf16_t)st[nt][i];
    __syncthreads();

    // O += P(16x64) x V(64x128); V^T read as B-fragments from global
#pragma unroll
    for (int ks2 = 0; ks2 < 2; ++ks2) {
      bf16x16 pa = load_fragA(&sP[wid * 16][ks2 * 32], 88);
#pragma unroll
      for (int nt = 0; nt < 8; ++nt) {
        bf16x16 vb = load_fragB(vp + (size_t)(nt * 16) * S_ + kk0 + ks2 * 32, S_);
        O[nt] = wmma_bf16(pa, vb, O[nt]);
      }
    }
    __syncthreads();
  }

  // normalize and store attn (bf16 [BS][H*128])
#pragma unroll
  for (int nt = 0; nt < 8; ++nt) {
    int d = h * 128 + nt * 16 + lcol;
#pragma unroll
    for (int i = 0; i < 8; ++i) {
      int row = wq0 + half * 8 + i;
      Ob[(size_t)(b * S_ + row) * (H_ * HD_) + d] = (bf16_t)(O[nt][i] / rsum[i]);
    }
  }
}

// ---------------- host launcher ----------------
extern "C" void kernel_launch(void* const* d_in, const int* in_sizes, int n_in,
                              void* d_out, int out_size, void* d_ws, size_t ws_size,
                              hipStream_t stream) {
  (void)in_sizes; (void)n_in; (void)out_size; (void)ws_size;
  const float* x = (const float*)d_in[0];
  const float* wq = (const float*)d_in[1];
  const float* wkv_a = (const float*)d_in[2];
  const float* gamma = (const float*)d_in[3];
  const float* wkv_b = (const float*)d_in[4];
  const float* wo = (const float*)d_in[5];
  const int* pos = (const int*)d_in[7];
  float* out = (float*)d_out;

  char* ws = (char*)d_ws;
  size_t o = 0;
  auto take = [&](size_t bytes) { size_t r = o; o += (bytes + 255) & ~(size_t)255; return r; };
  const size_t o_xb   = take((size_t)BS_ * D_ * 2);                 // x bf16
  const size_t o_wqT  = take((size_t)(H_ * 2 * HD_) * D_ * 2);      // wq^T
  const size_t o_waT  = take((size_t)(KVR_ + HD_) * D_ * 2);        // wkv_a^T
  const size_t o_wbT  = take((size_t)(KVH_ * 2 * HD_) * KVR_ * 2);  // wkv_b^T
  const size_t o_woT  = take((size_t)D_ * (H_ * HD_) * 2);          // wo^T
  const size_t o_qraw = take((size_t)BS_ * (H_ * 2 * HD_) * 2);     // q pre-rope bf16
  const size_t o_kvf  = take((size_t)BS_ * (KVR_ + HD_) * 4);       // kv f32
  const size_t o_qb   = take((size_t)B_ * H_ * S_ * 256 * 2);       // q [B][H][S][256]
  const size_t o_kvc  = take((size_t)BS_ * KVR_ * 2);               // normed latent
  const size_t o_kpe  = take((size_t)BS_ * HD_ * 2);                // roped k_pe
  const size_t o_kb   = take((size_t)B_ * KVH_ * S_ * 256 * 2);     // K
  const size_t o_vT   = take((size_t)B_ * KVH_ * HD_ * S_ * 2);     // V^T
  const size_t o_kvb  = o_xb;    // alias: kvb bf16 after x dead
  const size_t o_attn = o_qraw;  // alias: attn bf16 after qraw dead

  bf16_t* xb   = (bf16_t*)(ws + o_xb);
  bf16_t* wqT  = (bf16_t*)(ws + o_wqT);
  bf16_t* waT  = (bf16_t*)(ws + o_waT);
  bf16_t* wbT  = (bf16_t*)(ws + o_wbT);
  bf16_t* woT  = (bf16_t*)(ws + o_woT);
  bf16_t* qraw = (bf16_t*)(ws + o_qraw);
  float*  kvf  = (float*)(ws + o_kvf);
  bf16_t* qb   = (bf16_t*)(ws + o_qb);
  bf16_t* kvc  = (bf16_t*)(ws + o_kvc);
  bf16_t* kpe  = (bf16_t*)(ws + o_kpe);
  bf16_t* kb   = (bf16_t*)(ws + o_kb);
  bf16_t* vT   = (bf16_t*)(ws + o_vT);
  bf16_t* kvb  = (bf16_t*)(ws + o_kvb);
  bf16_t* attn = (bf16_t*)(ws + o_attn);

  const float scale = 1.0f / sqrtf((float)HD_);

  // 1. conversions (weights transposed to [N][K])
  {
    long long n = (long long)BS_ * D_;
    conv_bf16_kernel<<<(unsigned)((n + 255) / 256), 256, 0, stream>>>(x, xb, n);
  }
  transconv_kernel<<<(unsigned)(((long long)D_ * 4096 + 255) / 256), 256, 0, stream>>>(wq, wqT, D_, 4096);
  transconv_kernel<<<(unsigned)(((long long)D_ * 640 + 255) / 256), 256, 0, stream>>>(wkv_a, waT, D_, 640);
  transconv_kernel<<<(unsigned)(((long long)KVR_ * 2048 + 255) / 256), 256, 0, stream>>>(wkv_b, wbT, KVR_, 2048);
  transconv_kernel<<<(unsigned)(((long long)2048 * D_ + 255) / 256), 256, 0, stream>>>(wo, woT, 2048, D_);

  // 2. q = x @ wq (bf16 out)
  gemm_kernel<bf16_t><<<dim3(BS_ / 128, 4096 / 128), 256, 0, stream>>>(xb, wqT, qraw, BS_, 4096, D_);
  // 3. kv = x @ wkv_a (f32 out)
  gemm_kernel<float><<<dim3(BS_ / 128, 640 / 128), 256, 0, stream>>>(xb, waT, kvf, BS_, 640, D_);
  // 4. rope q + relayout
  ropeq_kernel<<<(BS_ * H_ * 128) / 256, 256, 0, stream>>>(qraw, pos, qb);
  // 5. layernorm + rope(k_pe)
  kvproc_kernel<<<BS_, 256, 0, stream>>>(kvf, gamma, pos, kvc, kpe);
  // 6. kvb = kv_c @ wkv_b (bf16 out, aliases x)
  gemm_kernel<bf16_t><<<dim3(BS_ / 128, 2048 / 128), 256, 0, stream>>>(kvc, wbT, kvb, BS_, 2048, KVR_);
  // 7. assemble K and V^T
  buildkv_kernel<<<(B_ * S_ * KVH_ * 128) / 256, 256, 0, stream>>>(kvb, kpe, kb, vT);
  // 8. flash attention (aliases qraw)
  attn_kernel<<<dim3(S_ / 128, H_, B_), 256, 0, stream>>>(qb, kb, vT, attn, scale);
  // 9. out = attn @ wo (f32 to d_out)
  gemm_kernel<float><<<dim3(BS_ / 128, D_ / 128), 256, 0, stream>>>(attn, woT, out, BS_, D_, H_ * HD_);
}
